// VDP_ViT_27745488732909
// MI455X (gfx1250) — matrix operand order repaired and checked by reading, not compiled, not run on hardware
//
#include <hip/hip_runtime.h>
#include <math.h>

// ---------------- problem constants ----------------
#define E_DIM   768
#define NTOK    4096          // B*S = 8*512
#define NE      (NTOK*E_DIM)  // 3145728
#define SEQ     512
#define NH      12
#define DH      64
#define BATCH   8
#define LOG_PRIOR_VAR (-4.605170185988091f)

typedef __attribute__((ext_vector_type(16))) __bf16 v16bf;
typedef __attribute__((ext_vector_type(8)))  float  v8f;

// ---------------- WMMA helpers ----------------
__device__ inline v8f wmma_bf16(v16bf a, v16bf b, v8f c) {
  // D = A(16x32 bf16) * B(32x16 bf16) + C(16x16 f32)
  return __builtin_amdgcn_wmma_f32_16x16x32_bf16(
      false, a, false, b, (short)0, c, false, false);
}

// A-fragment (16x32, M x K). ISA layout: lanes0-15 row=lane, K {k0..k0+7, k0+16..k0+23};
// lanes16-31 row=lane-16, K {k0+8..15, k0+24..31}. Two contiguous 16B loads.
__device__ inline v16bf load_a(const __bf16* base, int row_stride, int m0, int k0, int lane) {
  int l = lane & 15, hi = lane >> 4;
  const __bf16* p = base + (long)(m0 + l) * row_stride + k0 + hi * 8;
  v16bf r;
  *reinterpret_cast<uint4*>(&r)       = *reinterpret_cast<const uint4*>(p);
  *(reinterpret_cast<uint4*>(&r) + 1) = *reinterpret_cast<const uint4*>(p + 16);
  return r;
}

// B-fragment (32x16, K x N) from [N][K] row-major storage.
// lane n holds column n: lanes0-15 K {k0..k0+7, k0+16..k0+23}, lanes16-31 K {k0+8..15, k0+24..31}.
__device__ inline v16bf load_b(const __bf16* baseT, int kstride, int n0, int k0, int lane) {
  int l = lane & 15, hi = lane >> 4;
  const __bf16* p = baseT + (long)(n0 + l) * kstride + k0 + hi * 8;
  v16bf r;
  *reinterpret_cast<uint4*>(&r)       = *reinterpret_cast<const uint4*>(p);
  *(reinterpret_cast<uint4*>(&r) + 1) = *reinterpret_cast<const uint4*>(p + 16);
  return r;
}

__device__ inline float softplusf(float x) {
  return (x > 20.0f) ? x : log1pf(__expf(x));
}

// async 16B global -> LDS transfer (per active lane), tracked by ASYNCcnt
__device__ inline void async_copy_b128(const __bf16* lds_dst, const __bf16* gsrc) {
  unsigned lds_addr = (unsigned)(unsigned long long)(const void*)lds_dst;
  asm volatile("global_load_async_to_lds_b128 %0, %1, off"
               :: "v"(lds_addr), "v"(gsrc) : "memory");
}
__device__ inline void wait_asynccnt0() {
  asm volatile("s_wait_asynccnt 0x0" ::: "memory");
}

// ---------------- prep kernels ----------------
__global__ void zero_acc_kernel(float* acc) {
  if (threadIdx.x < 2) acc[threadIdx.x] = 0.0f;
}

// one block per token row: bf16 convert + row stats (xx = sum((mu/768)^2), ss = sum(sigma))
__global__ __launch_bounds__(256)
void prep_inputs_kernel(const float* __restrict__ mu, const float* __restrict__ sg,
                        __bf16* mubf, __bf16* sgbf, float* xx, float* ss) {
  int r = blockIdx.x, t = threadIdx.x;
  float px = 0.f, ps = 0.f;
  for (int i = 0; i < 3; ++i) {
    int c = t + 256 * i;
    float m = mu[(long)r * E_DIM + c];
    float s = sg[(long)r * E_DIM + c];
    mubf[(long)r * E_DIM + c] = (__bf16)m;
    sgbf[(long)r * E_DIM + c] = (__bf16)s;
    float xs = m * (1.0f / 768.0f);
    px += xs * xs;
    ps += s;
  }
  __shared__ float ra[256], rb[256];
  ra[t] = px; rb[t] = ps; __syncthreads();
  for (int st = 128; st > 0; st >>= 1) {
    if (t < st) { ra[t] += ra[t + st]; rb[t] += rb[t + st]; }
    __syncthreads();
  }
  if (t == 0) { xx[r] = ra[0]; ss[r] = rb[0]; }
}

// transpose weights to [N][K] bf16 + W^2/768, accumulate sum(mu^2) for KL
__global__ __launch_bounds__(256)
void prep_wt_kernel(const float* w0, const float* w1, const float* w2, const float* w3,
                    __bf16* Wt, __bf16* W2t, float* acc) {
  int n = blockIdx.x, w = blockIdx.y, t = threadIdx.x;
  const float* W = (w == 0) ? w0 : (w == 1) ? w1 : (w == 2) ? w2 : w3;
  long base = (long)w * E_DIM * E_DIM;
  float p = 0.f;
  for (int i = 0; i < 3; ++i) {
    int k = t + 256 * i;
    float v = W[(long)k * E_DIM + n];
    Wt[base + (long)n * E_DIM + k]  = (__bf16)v;
    W2t[base + (long)n * E_DIM + k] = (__bf16)(v * v * (1.0f / 768.0f));
    p += v * v;
  }
  __shared__ float ra[256];
  ra[t] = p; __syncthreads();
  for (int st = 128; st > 0; st >>= 1) {
    if (t < st) ra[t] += ra[t + st];
    __syncthreads();
  }
  if (t == 0) atomicAdd(&acc[1], ra[0]);
}

// softplus(w_sigma) + KL sigma-term accumulation
__global__ __launch_bounds__(256)
void prep_wsig_kernel(const float* s0, const float* s1, const float* s2, const float* s3,
                      float* sp_sig, float* acc) {
  int w = blockIdx.x, t = threadIdx.x;
  const float* S = (w == 0) ? s0 : (w == 1) ? s1 : (w == 2) ? s2 : s3;
  float p = 0.f;
  for (int i = 0; i < 3; ++i) {
    int j = t + 256 * i;
    float sig = S[j];
    float sp = softplusf(sig);
    sp_sig[w * E_DIM + j] = sp;
    p += (-sig + sp * 100.0f);   // /PRIOR_VAR
  }
  __shared__ float ra[256];
  ra[t] = p; __syncthreads();
  for (int st = 128; st > 0; st >>= 1) {
    if (t < st) ra[t] += ra[t + st];
    __syncthreads();
  }
  if (t == 0) atomicAdd(&acc[0], ra[0]);
}

// ---------------- fused VDP linear (GEMM) ----------------
// roles: 0=Q, 1=K, 2=V (bf16 head-layout outputs), 3=O (f32 to d_out)
// All 8 waves of a block share one 16-row A panel: stage it in LDS via
// ASYNCcnt-tracked global_load_async_to_lds_b128 (24KB mu + 24KB sigma,
// contiguous in memory), then stream B from global.
__global__ __launch_bounds__(256)
void gemm_kernel(const __bf16* __restrict__ Amu, const __bf16* __restrict__ Asg,
                 const __bf16* __restrict__ Wt,  const __bf16* __restrict__ W2t,
                 const float* __restrict__ spw, const float* __restrict__ xx,
                 const float* __restrict__ ss,
                 __bf16* o_mu, __bf16* o_sg, __bf16* o_mix, float* out_f32, int role) {
  __shared__ __bf16 amu_s[16 * E_DIM];   // 24KB
  __shared__ __bf16 asg_s[16 * E_DIM];   // 24KB

  int t = threadIdx.x;
  int lane = t & 31;
  int wid = t >> 5;
  int tm = blockIdx.x / 6;                 // 48 tn-tiles per tm, 8 per block
  int tn = (blockIdx.x % 6) * 8 + wid;
  int r0 = tm * 16, c0 = tn * 16;

  // async-stage A panel: 16 rows x 768 cols bf16, rows contiguous => flat 24KB copy
  {
    const __bf16* gmu = Amu + (long)r0 * E_DIM;
    const __bf16* gsg = Asg + (long)r0 * E_DIM;
    for (int i = 0; i < 6; ++i) {
      int eoff = (t + i * 256) * 8;        // 8 bf16 = 16B per lane-transfer
      async_copy_b128(&amu_s[eoff], gmu + eoff);
      async_copy_b128(&asg_s[eoff], gsg + eoff);
    }
    wait_asynccnt0();
    __syncthreads();
  }

  v8f accm = {}, accs = {};
  for (int kt = 0; kt < 24; ++kt) {
    int k0 = kt * 32;
    if (kt + 1 < 24) {
      __builtin_prefetch(Wt  + (long)(c0 + (lane & 15)) * E_DIM + k0 + 32, 0, 0);
      __builtin_prefetch(W2t + (long)(c0 + (lane & 15)) * E_DIM + k0 + 32, 0, 0);
    }
    v16bf am = load_a(amu_s, E_DIM, 0, k0, lane);   // ds_read_b128 x2
    v16bf as = load_a(asg_s, E_DIM, 0, k0, lane);
    v16bf bw = load_b(Wt,  E_DIM, c0, k0, lane);    // global_load_b128 x2
    v16bf b2 = load_b(W2t, E_DIM, c0, k0, lane);
    accm = wmma_bf16(am, bw, accm);
    accs = wmma_bf16(as, b2, accs);
  }
  int l = lane & 15, hi = lane >> 4;
  for (int j = 0; j < 8; ++j) {
    int r = r0 + j + hi * 8;
    int c = c0 + l;
    float spc = spw[c];
    float mu = accm[j];
    float sg = softplusf(accs[j] + xx[r] * spc + ss[r] * spc * (1.0f / 768.0f));
    if (role == 3) {
      out_f32[(long)r * E_DIM + c] = mu;
      out_f32[(long)NE + (long)r * E_DIM + c] = sg;
    } else {
      int b = r >> 9, s = r & 511, h = c >> 6, dd = c & 63;
      long off = (role == 2) ? (((long)(b * NH + h) * DH + dd) * SEQ + s)   // transposed [d][S] (B-operand)
                             : (((long)(b * NH + h) * SEQ + s) * DH + dd);  // direct [S][d]
      o_mu[off] = (__bf16)mu;
      o_sg[off] = (__bf16)sg;
      // Q: mu^2/dk ; K: (mu^2+sg)/dk ; V: (mu^2+sg)/wd
      float mix = (mu * mu + ((role == 0) ? 0.0f : sg)) *
                  ((role == 2) ? (1.0f / 512.0f) : (1.0f / 64.0f));
      o_mix[off] = (__bf16)mix;
    }
  }
}

// ---------------- fused attention ----------------
// 1 block = (b, h, 32-row q block); 512 threads = 16 waves.
// LDS: mu_score f32[32][512] + sg_score f32[32][512] + mu_w bf16[32][512] + Sigma_w bf16[32][512] = 192KB
__global__ __launch_bounds__(512)
void attn_kernel(const __bf16* __restrict__ muq, const __bf16* __restrict__ mu2q,
                 const __bf16* __restrict__ sgq,
                 const __bf16* __restrict__ muk, const __bf16* __restrict__ sgk,
                 const __bf16* __restrict__ kmix,
                 const __bf16* __restrict__ muvT, const __bf16* __restrict__ sgvT,
                 const __bf16* __restrict__ vmixT,
                 __bf16* mucat, __bf16* sgcat) {
  extern __shared__ char smem[];
  float*  mu_s = (float*)smem;                 // 64KB
  float*  sg_s = mu_s + 32 * 512;              // 64KB
  __bf16* muw  = (__bf16*)(sg_s + 32 * 512);   // 32KB
  __bf16* sgw  = muw + 32 * 512;               // 32KB

  int tid = threadIdx.x, lane = tid & 31, wave = tid >> 5;
  int l = lane & 15, hi = lane >> 4;
  int bid = blockIdx.x;
  int qb = bid & 15;
  int h  = (bid >> 4) % NH;
  int b  = bid / (16 * NH);
  long head = (long)(b * NH + h) * SEQ * DH;

  const __bf16* qm = muq  + head + (long)qb * 32 * DH;
  const __bf16* q2 = mu2q + head + (long)qb * 32 * DH;
  const __bf16* qs = sgq  + head + (long)qb * 32 * DH;
  const __bf16* km = muk  + head;
  const __bf16* ks = sgk  + head;
  const __bf16* kx = kmix + head;

  // ---- phase 1: scores. 64 tiles (2 x 32), 4 per wave.
  int tq = wave >> 3;
  int tkbase = (wave & 7) * 4;
  v8f accm[4] = {{0}}; v8f accs[4] = {{0}};
  for (int kk = 0; kk < 2; ++kk) {
    int k0 = kk * 32;
    v16bf amu = load_a(qm, DH, tq * 16, k0, lane);
    v16bf am2 = load_a(q2, DH, tq * 16, k0, lane);
    v16bf asg = load_a(qs, DH, tq * 16, k0, lane);
    for (int i = 0; i < 4; ++i) {
      int n0 = (tkbase + i) * 16;
      v16bf bm = load_b(km, DH, n0, k0, lane);
      accm[i] = wmma_bf16(amu, bm, accm[i]);          // mu_score
      v16bf bs = load_b(ks, DH, n0, k0, lane);
      accs[i] = wmma_bf16(am2, bs, accs[i]);          // a = (mu_q^2/dk) sg_k^T
      v16bf bx = load_b(kx, DH, n0, k0, lane);
      accs[i] = wmma_bf16(asg, bx, accs[i]);          // b^T + c = sg_q ((mu_k^2+sg_k)/dk)^T
    }
  }
  for (int i = 0; i < 4; ++i) {
    int col = (tkbase + i) * 16 + l;
    for (int j = 0; j < 8; ++j) {
      int row = tq * 16 + j + hi * 8;
      mu_s[row * 512 + col] = accm[i][j];
      sg_s[row * 512 + col] = accs[i][j];
    }
  }
  __syncthreads();

  // ---- phase 2: softmax + Sigma_w. one wave per 2 rows.
  for (int rr = 0; rr < 2; ++rr) {
    int row = wave * 2 + rr;
    float ev[16];
    float vmax = -1e30f;
    for (int i = 0; i < 16; ++i) {
      float v = mu_s[row * 512 + lane + 32 * i] * 0.125f;   // / sqrt(dk)
      ev[i] = v;
      vmax = fmaxf(vmax, v);
    }
    for (int m = 16; m >= 1; m >>= 1) vmax = fmaxf(vmax, __shfl_xor(vmax, m, 32));
    float sum = 0.f;
    for (int i = 0; i < 16; ++i) { ev[i] = __expf(ev[i] - vmax); sum += ev[i]; }
    for (int m = 16; m >= 1; m >>= 1) sum += __shfl_xor(sum, m, 32);
    float inv = 1.0f / sum;
    for (int i = 0; i < 16; ++i) {
      int idx = row * 512 + lane + 32 * i;
      float w = ev[i] * inv;
      float g = w - w * w; g = g * g;
      float Sw = g * sg_s[idx] * (1.0f / 64.0f) * (1.0f / 512.0f);  // grad*sigma_scaled/wd
      muw[idx] = (__bf16)w;
      sgw[idx] = (__bf16)Sw;
    }
  }
  __syncthreads();

  // ---- phase 3: output GEMMs. 8 tiles (2 x 4), waves 0..7.
  if (wave < 8) {
    int tq2 = wave >> 2;
    int td  = wave & 3;
    const __bf16* vm = muvT  + head;
    const __bf16* vs = sgvT  + head;
    const __bf16* vx = vmixT + head;
    v8f am = {}, ad = {}, ae = {};
    for (int it = 0; it < 16; ++it) {
      int k0 = it * 32;
      v16bf aw = load_a(muw, 512, tq2 * 16, k0, lane);
      v16bf as = load_a(sgw, 512, tq2 * 16, k0, lane);
      v16bf aw2;
      for (int e2 = 0; e2 < 16; ++e2) {
        float x = (float)aw[e2];
        aw2[e2] = (__bf16)(x * x * (1.0f / 512.0f));        // mu_w^2/wd
      }
      v16bf bm = load_b(vm, 512, td * 16, k0, lane);
      am = wmma_bf16(aw, bm, am);                           // mu_out
      v16bf bs = load_b(vs, 512, td * 16, k0, lane);
      ad = wmma_bf16(aw2, bs, ad);                          // d
      v16bf bx = load_b(vx, 512, td * 16, k0, lane);
      ae = wmma_bf16(as, bx, ae);                           // e + f
    }
    for (int j = 0; j < 8; ++j) {
      int rloc = tq2 * 16 + j + hi * 8;
      int r = b * SEQ + qb * 32 + rloc;
      int c = h * DH + td * 16 + l;
      mucat[(long)r * E_DIM + c] = (__bf16)am[j];
      float sg2 = softplusf(ad[j] + ae[j]);
      sgcat[(long)r * E_DIM + c] = (__bf16)sg2;
    }
  }
}

// row stats of concatenated attention outputs (bf16 in)
__global__ __launch_bounds__(256)
void cat_stats_kernel(const __bf16* __restrict__ mubf, const __bf16* __restrict__ sgbf,
                      float* xx, float* ss) {
  int r = blockIdx.x, t = threadIdx.x;
  float px = 0.f, ps = 0.f;
  for (int i = 0; i < 3; ++i) {
    int c = t + 256 * i;
    float m = (float)mubf[(long)r * E_DIM + c];
    float s = (float)sgbf[(long)r * E_DIM + c];
    float xs = m * (1.0f / 768.0f);
    px += xs * xs;
    ps += s;
  }
  __shared__ float ra[256], rb[256];
  ra[t] = px; rb[t] = ps; __syncthreads();
  for (int st = 128; st > 0; st >>= 1) {
    if (t < st) { ra[t] += ra[t + st]; rb[t] += rb[t + st]; }
    __syncthreads();
  }
  if (t == 0) { xx[r] = ra[0]; ss[r] = rb[0]; }
}

__global__ void finalize_kl_kernel(const float* acc, float* out) {
  if (threadIdx.x == 0 && blockIdx.x == 0) {
    float kl = 0.5f * (4.0f * (LOG_PRIOR_VAR - 1.0f)
                       + acc[0] * (1.0f / 768.0f)
                       + acc[1] * (100.0f / 589824.0f));
    out[2L * NE] = kl;
  }
}

// ---------------- host launcher ----------------
extern "C" void kernel_launch(void* const* d_in, const int* in_sizes, int n_in,
                              void* d_out, int out_size, void* d_ws, size_t ws_size,
                              hipStream_t stream) {
  (void)in_sizes; (void)n_in; (void)out_size; (void)ws_size;

  const float* mu_in = (const float*)d_in[0];
  const float* sg_in = (const float*)d_in[1];
  const float* wq_mu = (const float*)d_in[2];
  const float* wq_sg = (const float*)d_in[3];
  const float* wk_mu = (const float*)d_in[4];
  const float* wk_sg = (const float*)d_in[5];
  const float* wv_mu = (const float*)d_in[6];
  const float* wv_sg = (const float*)d_in[7];
  const float* wo_mu = (const float*)d_in[8];
  const float* wo_sg = (const float*)d_in[9];
  float* out = (float*)d_out;

  // workspace carve-up (all offsets 256B-aligned)
  char* ws = (char*)d_ws;
  size_t off = 0;
  auto take = [&](size_t bytes) {
    char* p = ws + off;
    off += (bytes + 255) & ~(size_t)255;
    return p;
  };
  const size_t NEbf = (size_t)NE * 2;           // bf16 [4096,768]
  const size_t Wbf  = (size_t)4 * E_DIM * E_DIM * 2;

  __bf16* mu_in_bf = (__bf16*)take(NEbf);
  __bf16* sg_in_bf = (__bf16*)take(NEbf);
  float*  xx_in    = (float*)take(NTOK * 4);
  float*  ss_in    = (float*)take(NTOK * 4);
  __bf16* Wt_bf    = (__bf16*)take(Wbf);
  __bf16* W2t_bf   = (__bf16*)take(Wbf);
  float*  sp_sig   = (float*)take(4 * E_DIM * 4);
  float*  klacc    = (float*)take(256);
  __bf16* muq  = (__bf16*)take(NEbf);
  __bf16* mu2q = (__bf16*)take(NEbf);
  __bf16* sgq  = (__bf16*)take(NEbf);
  __bf16* muk  = (__bf16*)take(NEbf);
  __bf16* sgk  = (__bf16*)take(NEbf);
  __bf16* kmix = (__bf16*)take(NEbf);
  __bf16* muvT = (__bf16*)take(NEbf);
  __bf16* sgvT = (__bf16*)take(NEbf);
  __bf16* vmixT = (__bf16*)take(NEbf);
  __bf16* mucat = (__bf16*)take(NEbf);
  __bf16* sgcat = (__bf16*)take(NEbf);
  float*  xx_cat = (float*)take(NTOK * 4);
  float*  ss_cat = (float*)take(NTOK * 4);

  const long WW = (long)E_DIM * E_DIM;

  zero_acc_kernel<<<1, 64, 0, stream>>>(klacc);
  prep_inputs_kernel<<<NTOK, 256, 0, stream>>>(mu_in, sg_in, mu_in_bf, sg_in_bf, xx_in, ss_in);
  prep_wt_kernel<<<dim3(E_DIM, 4), 256, 0, stream>>>(wq_mu, wk_mu, wv_mu, wo_mu, Wt_bf, W2t_bf, klacc);
  prep_wsig_kernel<<<4, 256, 0, stream>>>(wq_sg, wk_sg, wv_sg, wo_sg, sp_sig, klacc);

  // Q, K, V projections (12288 16x16 tiles, 8 waves/block, async-staged A panel)
  gemm_kernel<<<1536, 256, 0, stream>>>(mu_in_bf, sg_in_bf, Wt_bf + 0 * WW, W2t_bf + 0 * WW,
                                        sp_sig + 0 * E_DIM, xx_in, ss_in,
                                        muq, sgq, mu2q, nullptr, 0);
  gemm_kernel<<<1536, 256, 0, stream>>>(mu_in_bf, sg_in_bf, Wt_bf + 1 * WW, W2t_bf + 1 * WW,
                                        sp_sig + 1 * E_DIM, xx_in, ss_in,
                                        muk, sgk, kmix, nullptr, 1);
  gemm_kernel<<<1536, 256, 0, stream>>>(mu_in_bf, sg_in_bf, Wt_bf + 2 * WW, W2t_bf + 2 * WW,
                                        sp_sig + 2 * E_DIM, xx_in, ss_in,
                                        muvT, sgvT, vmixT, nullptr, 2);

  // fused attention: scores -> softmax -> Sigma_w -> output GEMMs, all in LDS
  const int attn_lds = 192 * 1024;
  hipFuncSetAttribute(reinterpret_cast<const void*>(attn_kernel),
                      hipFuncAttributeMaxDynamicSharedMemorySize, attn_lds);
  attn_kernel<<<BATCH * NH * 16, 512, attn_lds, stream>>>(
      muq, mu2q, sgq, muk, sgk, kmix, muvT, sgvT, vmixT, mucat, sgcat);

  cat_stats_kernel<<<NTOK, 256, 0, stream>>>(mucat, sgcat, xx_cat, ss_cat);

  // O projection writes final mu_out / Sigma_out
  gemm_kernel<<<1536, 256, 0, stream>>>(mucat, sgcat, Wt_bf + 3 * WW, W2t_bf + 3 * WW,
                                        sp_sig + 3 * E_DIM, xx_cat, ss_cat,
                                        nullptr, nullptr, nullptr, out, 3);

  finalize_kl_kernel<<<1, 64, 0, stream>>>(klacc, out);
}
